// FieldAttentionModule_893353198411
// MI455X (gfx1250) — matrix operand order repaired and checked by reading, not compiled, not run on hardware
//
#include <hip/hip_runtime.h>
#include <hip/hip_bf16.h>

// ---------------------------------------------------------------------------
// FieldAttention on MI455X (gfx1250): bf16 WMMA everywhere, f32 accumulate.
// W=256 windows, Sq=256, Skv=480 (halo), F=256, ATT=128, NH=4, NHC=32.
// ---------------------------------------------------------------------------

#define DEV static __device__ __forceinline__

typedef float  v8f   __attribute__((ext_vector_type(8)));
typedef __bf16 bfx16 __attribute__((ext_vector_type(16)));

union FragCast { uint4 u[2]; bfx16 v; };

DEV unsigned lane_id() { return threadIdx.x & 31u; }

// f32 -> bf16 (round-to-nearest-even), stored as u16
DEV unsigned short f2bf(float f) {
    unsigned u = __float_as_uint(f);
    u = (u + 0x7FFFu + ((u >> 16) & 1u)) >> 16;
    return (unsigned short)u;
}

// A fragment 16x32 bf16, row-major source, row stride ld (elements).
// lane<16: row=lane, halves 0..7 = K0..7, 8..15 = K16..23
// lane>=16: row=lane-16, halves 0..7 = K8..15, 8..15 = K24..31
DEV bfx16 frag_a(const unsigned short* base, int ld) {
    unsigned ln = lane_id();
    const unsigned short* p = base + (ln & 15u) * ld + ((ln & 16u) ? 8 : 0);
    FragCast f;
    f.u[0] = *(const uint4*)(p);
    f.u[1] = *(const uint4*)(p + 16);
    return f.v;
}

// B fragment 32x16 bf16 from a TRANSPOSED store [N][K], row stride ld.
// lane<16: col=lane, halves 0..7 = K0..7, 8..15 = K16..23
// lane>=16: col=lane-16, halves 0..7 = K8..15, 8..15 = K24..31
DEV bfx16 frag_b(const unsigned short* bT, int ld) {
    unsigned ln = lane_id();
    const unsigned short* p = bT + (ln & 15u) * ld + ((ln & 16u) ? 8 : 0);
    FragCast f;
    f.u[0] = *(const uint4*)(p);
    f.u[1] = *(const uint4*)(p + 16);
    return f.v;
}

DEV v8f wmma_bf16(bfx16 a, bfx16 b, v8f c) {
    // (neg_a, A, neg_b, B, c_mod, C, reuse_a, reuse_b)
    return __builtin_amdgcn_wmma_f32_16x16x32_bf16(false, a, false, b,
                                                   (short)0, c, false, false);
}

DEV v8f v8f_zero() {
    v8f z;
#pragma unroll
    for (int i = 0; i < 8; ++i) z[i] = 0.f;
    return z;
}

// ---------------------------------------------------------------------------
// Kernel 1: q = x @ Wq  -> q_ws[w][h][sq][c] bf16
// grid = W*16 (16-row tiles), block = 128 (4 waves, 2 N-tiles each, N=128)
// ---------------------------------------------------------------------------
__global__ __launch_bounds__(128)
void k_proj_q(const float* __restrict__ x, const float* __restrict__ Wq,
              unsigned short* __restrict__ qws) {
    __shared__ unsigned short xs[16 * 256];    // A tile, bf16
    __shared__ unsigned short wqTc[128 * 32];  // Wq chunk, transposed [N][Kc]
    const int tid = threadIdx.x;
    const int w = blockIdx.x >> 4, mt = blockIdx.x & 15;

    const float* xrow = x + ((size_t)w * 256 + (size_t)mt * 16) * 256;
    for (int i = tid; i < 16 * 256; i += 128) xs[i] = f2bf(xrow[i]);
    __syncthreads();

    bfx16 afr[8];
#pragma unroll
    for (int kc = 0; kc < 8; ++kc) afr[kc] = frag_a(xs + kc * 32, 256);

    const int wid = tid >> 5;
    v8f accs[2] = { v8f_zero(), v8f_zero() };

    for (int kc = 0; kc < 8; ++kc) {
        __syncthreads();
        for (int i = tid; i < 128 * 32; i += 128) {
            int kk = i >> 7, n = i & 127;                       // coalesced read
            wqTc[n * 32 + kk] = f2bf(Wq[(kc * 32 + kk) * 128 + n]);
        }
        __syncthreads();
        accs[0] = wmma_bf16(afr[kc], frag_b(wqTc + (wid * 2 + 0) * 512, 32), accs[0]);
        accs[1] = wmma_bf16(afr[kc], frag_b(wqTc + (wid * 2 + 1) * 512, 32), accs[1]);
    }

    const unsigned ln = lane_id();
    const int mb = (ln & 16u) ? 8 : 0;
#pragma unroll
    for (int i = 0; i < 2; ++i) {
        int nt = wid * 2 + i;
        int a = nt * 16 + (int)(ln & 15u);
        int h = a >> 5, c = a & 31;
        unsigned short* qd = qws + (((size_t)w * 4 + h) * 256) * 32 + c;
#pragma unroll
        for (int r = 0; r < 8; ++r) {
            int sq = mt * 16 + mb + r;
            qd[(size_t)sq * 32] = f2bf(accs[i][r]);
        }
    }
}

// ---------------------------------------------------------------------------
// Kernel 2: kv = gather_halo(x) @ Wkv + b  -> k_ws/v_ws[w][h][skv][c] bf16
// grid = W*30 (30 row-tiles of 480), block = 256 (8 waves, 2 N-tiles, N=256)
// ---------------------------------------------------------------------------
__global__ __launch_bounds__(256)
void k_proj_kv(const float* __restrict__ x, const float* __restrict__ Wkv,
               const float* __restrict__ bkv,
               unsigned short* __restrict__ kws, unsigned short* __restrict__ vws) {
    __shared__ unsigned short xs[16 * 256];
    __shared__ unsigned short wkTc[256 * 32];
    const int tid = threadIdx.x;
    const int w = blockIdx.x / 30, st = blockIdx.x % 30;
    const int vv = (w >> 7) & 1, t = (w >> 6) & 1, n = (w >> 1) & 31, d = w & 1;

    // halo gather (time along Tt: 8->10, depth along Dd: 4->6), zero-filled OOB
    for (int i = tid; i < 16 * 256; i += 256) {
        int r = i >> 8, f = i & 255;
        int skv = st * 16 + r;
        int tt = skv / 48, rm = skv % 48, nn = rm / 6, dd = rm % 6;
        int ts = t, tts, ds = d, dds;
        if (tt == 0)      { ts = t - 1; tts = 7; }
        else if (tt == 9) { ts = t + 1; tts = 0; }
        else              { tts = tt - 1; }
        if (dd == 0)      { ds = d - 1; dds = 3; }
        else if (dd == 5) { ds = d + 1; dds = 0; }
        else              { dds = dd - 1; }
        unsigned short val = 0;
        if (ts >= 0 && ts < 2 && ds >= 0 && ds < 2) {
            int wsrc = vv * 128 + ts * 64 + n * 2 + ds;
            int sqs  = tts * 32 + nn * 4 + dds;
            val = f2bf(x[((size_t)wsrc * 256 + sqs) * 256 + f]);
        }
        xs[i] = val;
    }
    __syncthreads();

    bfx16 afr[8];
#pragma unroll
    for (int kc = 0; kc < 8; ++kc) afr[kc] = frag_a(xs + kc * 32, 256);

    const int wid = tid >> 5;
    v8f accs[2] = { v8f_zero(), v8f_zero() };

    for (int kc = 0; kc < 8; ++kc) {
        __syncthreads();
        for (int i = tid; i < 256 * 32; i += 256) {
            int kk = i >> 8, nc = i & 255;
            wkTc[nc * 32 + kk] = f2bf(Wkv[(kc * 32 + kk) * 256 + nc]);
        }
        __syncthreads();
        accs[0] = wmma_bf16(afr[kc], frag_b(wkTc + (wid * 2 + 0) * 512, 32), accs[0]);
        accs[1] = wmma_bf16(afr[kc], frag_b(wkTc + (wid * 2 + 1) * 512, 32), accs[1]);
    }

    const unsigned ln = lane_id();
    const int mb = (ln & 16u) ? 8 : 0;
#pragma unroll
    for (int i = 0; i < 2; ++i) {
        int nt = wid * 2 + i;
        int a = nt * 16 + (int)(ln & 15u);    // column in [0,256)
        int h = a >> 6, rr = a & 63;
        float bias = bkv[a];
#pragma unroll
        for (int r = 0; r < 8; ++r) {
            int skv = st * 16 + mb + r;
            float val = accs[i][r] + bias;
            size_t rowb = (((size_t)w * 4 + h) * 480 + skv) * 32;
            if (rr < 32) kws[rowb + rr]        = f2bf(val);
            else         vws[rowb + (rr - 32)] = f2bf(val);
        }
    }
}

// ---------------------------------------------------------------------------
// Kernel 3: flash attention per (w,h).
// grid = W*NH = 1024, block = 512 (16 waves, one 16-row q tile per wave)
// kv streamed in two LDS-resident halves (256 + 224 rows), chunks of 32 cols.
// ---------------------------------------------------------------------------
__global__ __launch_bounds__(512)
void k_attn(const unsigned short* __restrict__ qws,
            const unsigned short* __restrict__ kws,
            const unsigned short* __restrict__ vws,
            unsigned short* __restrict__ ows) {
    __shared__ unsigned short ks[256 * 32];    // k half   [j][c]
    __shared__ unsigned short vts[32 * 256];   // v half^T [c][j]
    __shared__ unsigned short pbuf[16 * 512];  // per-wave P scratch 16x32

    const int tid = threadIdx.x, wid = tid >> 5;
    const unsigned ln = lane_id();
    const int w = blockIdx.x >> 2, h = blockIdx.x & 3;
    const size_t wh = (size_t)w * 4 + h;
    const float SC = 0.17677669529663687f;     // 1/sqrt(32)

    bfx16 aq = frag_a(qws + (wh * 256 + (size_t)wid * 16) * 32, 32);

    v8f o0 = v8f_zero(), o1 = v8f_zero();
    float m[8], l[8];
#pragma unroll
    for (int r = 0; r < 8; ++r) { m[r] = -3.0e38f; l[r] = 0.f; }

    unsigned short* pw = pbuf + wid * 512;
    const int mb = (ln & 16u) ? 8 : 0;
    const int col = (int)(ln & 15u);

    for (int half = 0; half < 2; ++half) {
        const int j0 = half * 256;
        const int cnt = half ? 224 : 256;
        __syncthreads();
        {   // stage K (coalesced u32 copies)
            const unsigned int* src = (const unsigned int*)(kws + (wh * 480 + j0) * 32);
            unsigned int* dst = (unsigned int*)ks;
            for (int i = tid; i < cnt * 16; i += 512) dst[i] = src[i];
        }
        {   // stage V transposed: vts[c][j]
            const unsigned short* vsrc = vws + (wh * 480 + j0) * 32;
            for (int i = tid; i < cnt * 32; i += 512) {
                int s = i >> 5, c = i & 31;
                vts[c * 256 + s] = vsrc[i];
            }
        }
        __syncthreads();

        const int nch = cnt >> 5;              // 8 then 7 chunks of 32
        for (int jc = 0; jc < nch; ++jc) {
            const int jl = jc * 32;
            v8f s0 = wmma_bf16(aq, frag_b(ks + (jl +  0) * 32, 32), v8f_zero());
            v8f s1 = wmma_bf16(aq, frag_b(ks + (jl + 16) * 32, 32), v8f_zero());

#pragma unroll
            for (int r = 0; r < 8; ++r) {
                float a0 = s0[r] * SC, a1 = s1[r] * SC;
                float mx = fmaxf(a0, a1);
                mx = fmaxf(mx, __shfl_xor(mx, 1, 32));
                mx = fmaxf(mx, __shfl_xor(mx, 2, 32));
                mx = fmaxf(mx, __shfl_xor(mx, 4, 32));
                mx = fmaxf(mx, __shfl_xor(mx, 8, 32));
                float nm = fmaxf(m[r], mx);
                float al = __expf(m[r] - nm);
                float p0 = __expf(a0 - nm), p1 = __expf(a1 - nm);
                float sm = p0 + p1;
                sm += __shfl_xor(sm, 1, 32);
                sm += __shfl_xor(sm, 2, 32);
                sm += __shfl_xor(sm, 4, 32);
                sm += __shfl_xor(sm, 8, 32);
                l[r] = l[r] * al + sm;
                m[r] = nm;
                o0[r] *= al;
                o1[r] *= al;
                int row = mb + r;
                pw[row * 32 + col]      = f2bf(p0);
                pw[row * 32 + 16 + col] = f2bf(p1);
            }
            // per-wave LDS RAW: LDS is in-order within a wave; stop compiler
            // from hoisting the fragment reads above the P stores.
            __builtin_amdgcn_wave_barrier();

            bfx16 pa = frag_a(pw, 32);
            o0 = wmma_bf16(pa, frag_b(vts +  0 * 256 + jl, 256), o0);
            o1 = wmma_bf16(pa, frag_b(vts + 16 * 256 + jl, 256), o1);
        }
    }

#pragma unroll
    for (int r = 0; r < 8; ++r) {
        int sq = wid * 16 + mb + r;
        float inv = 1.0f / l[r];
        unsigned short* od = ows + ((size_t)w * 256 + sq) * 128 + h * 32;
        od[col]      = f2bf(o0[r] * inv);
        od[16 + col] = f2bf(o1[r] * inv);
    }
}

// ---------------------------------------------------------------------------
// Kernel 4: out = x + gamma * (o @ Wo).  K=128 (4 chunks), N=F=256.
// grid = W*16, block = 256 (8 waves, 2 N-tiles each)
// ---------------------------------------------------------------------------
__global__ __launch_bounds__(256)
void k_out(const float* __restrict__ x, const unsigned short* __restrict__ ows,
           const float* __restrict__ Wo, const float* __restrict__ gamma,
           float* __restrict__ out) {
    __shared__ unsigned short os[16 * 128];
    __shared__ unsigned short woTc[256 * 32];
    const int tid = threadIdx.x;
    const int w = blockIdx.x >> 4, mt = blockIdx.x & 15;

    const float* xt = x + ((size_t)w * 256 + (size_t)mt * 16) * 256;
    if (tid < 16) __builtin_prefetch(xt + tid * 256, 0, 3);

    {   // stage o tile (bf16, u32 copies)
        const unsigned int* s = (const unsigned int*)(ows + ((size_t)w * 256 + (size_t)mt * 16) * 128);
        unsigned int* dcp = (unsigned int*)os;
        for (int i = tid; i < 1024; i += 256) dcp[i] = s[i];
    }
    __syncthreads();

    bfx16 afr[4];
#pragma unroll
    for (int kc = 0; kc < 4; ++kc) afr[kc] = frag_a(os + kc * 32, 128);

    const int wid = tid >> 5;
    v8f accs[2] = { v8f_zero(), v8f_zero() };

    for (int kc = 0; kc < 4; ++kc) {
        __syncthreads();
        for (int i = tid; i < 256 * 32; i += 256) {
            int kk = i >> 8, f = i & 255;
            woTc[f * 32 + kk] = f2bf(Wo[(kc * 32 + kk) * 256 + f]);
        }
        __syncthreads();
        accs[0] = wmma_bf16(afr[kc], frag_b(woTc + (wid * 2 + 0) * 512, 32), accs[0]);
        accs[1] = wmma_bf16(afr[kc], frag_b(woTc + (wid * 2 + 1) * 512, 32), accs[1]);
    }

    const unsigned ln = lane_id();
    const int mb = (ln & 16u) ? 8 : 0;
#pragma unroll
    for (int i = 0; i < 2; ++i) {
        int nt = wid * 2 + i;
        int f = nt * 16 + (int)(ln & 15u);
        float g = gamma[f];
#pragma unroll
        for (int r = 0; r < 8; ++r) {
            int sq = mt * 16 + mb + r;
            size_t idx = ((size_t)w * 256 + sq) * 256 + f;
            out[idx] = x[idx] + g * accs[i][r];
        }
    }
}

// ---------------------------------------------------------------------------
extern "C" void kernel_launch(void* const* d_in, const int* in_sizes, int n_in,
                              void* d_out, int out_size, void* d_ws, size_t ws_size,
                              hipStream_t stream) {
    (void)in_sizes; (void)n_in; (void)out_size; (void)ws_size;
    const float* x     = (const float*)d_in[0];
    const float* Wq    = (const float*)d_in[1];
    const float* Wkv   = (const float*)d_in[2];
    const float* bkv   = (const float*)d_in[3];
    const float* Wo    = (const float*)d_in[4];
    const float* gamma = (const float*)d_in[5];
    float* out = (float*)d_out;

    // workspace layout (bf16 as u16):
    //   q:  W*NH*Sq*NHC  = 8,388,608  elems -> 16,777,216 B
    //   k:  W*NH*Skv*NHC = 15,728,640 elems -> 31,457,280 B
    //   v:  same as k
    //   o:  W*Sq*ATT     = 8,388,608  elems -> 16,777,216 B
    char* ws = (char*)d_ws;
    unsigned short* qws = (unsigned short*)(ws);
    unsigned short* kws = (unsigned short*)(ws + 16777216u);
    unsigned short* vws = (unsigned short*)(ws + 16777216u + 31457280u);
    unsigned short* ows = (unsigned short*)(ws + 16777216u + 2u * 31457280u);

    k_proj_q <<<dim3(256 * 16), dim3(128), 0, stream>>>(x, Wq, qws);
    k_proj_kv<<<dim3(256 * 30), dim3(256), 0, stream>>>(x, Wkv, bkv, kws, vws);
    k_attn   <<<dim3(256 * 4),  dim3(512), 0, stream>>>(qws, kws, vws, ows);
    k_out    <<<dim3(256 * 16), dim3(256), 0, stream>>>(x, ows, Wo, gamma, out);
}
